// GatedTransformerXLLayer_55336358642568
// MI455X (gfx1250) — matrix operand order repaired and checked by reading, not compiled, not run on hardware
//
#include <hip/hip_runtime.h>

// ---------------------------------------------------------------------------
// Gated Transformer-XL layer for MI455X (gfx1250), wave32 + WMMA bf16.
// - All GEMMs use v_wmma_f32_16x16x32_bf16, 128x128 tiles, 8 waves/block.
// - Tiles staged via gfx1250 async global->LDS copies (ASYNCcnt) with weights
//   pre-transposed to [N][K] so both operands are row-contiguous.
// - Fused relative attention per (batch, head, 16-query strip).
// ---------------------------------------------------------------------------

#define CURQ 512
#define MEMR 512
#define BSZ  4
#define DIM  1024
#define NH   16
#define HDIM 64
#define HID  4096
#define FULL 1024
#define ATT_SCALE 0.125f

typedef unsigned short u16;
typedef unsigned int   u32;

typedef __attribute__((ext_vector_type(16))) __bf16 v16bf;
typedef __attribute__((ext_vector_type(8)))  float  v8f;
typedef __attribute__((ext_vector_type(4)))  u32    u32x4;

union Frag {
    v16bf v;
    u32x4 q[2];
    u16   u[16];
};

__device__ __forceinline__ v8f vzero() {
    v8f z;
#pragma unroll
    for (int i = 0; i < 8; ++i) z[i] = 0.0f;
    return z;
}

__device__ __forceinline__ u16 f2bf(float f) {
    union { float f; u32 u; } c; c.f = f;
    u32 u = c.u;
    if ((u & 0x7fffffffu) > 0x7f800000u) return (u16)((u >> 16) | 0x40); // NaN
    u32 r = u + 0x7fffu + ((u >> 16) & 1u);                              // RNE
    return (u16)(r >> 16);
}

__device__ __forceinline__ float bf2f(u16 h) {
    union { u32 u; float f; } c; c.u = ((u32)h) << 16; return c.f;
}

__device__ __forceinline__ v8f wmma_bf16(const Frag& a, const Frag& b, v8f c) {
    return __builtin_amdgcn_wmma_f32_16x16x32_bf16(
        false, a.v, false, b.v, (short)0, c, false, false);
}

// gfx1250 async global->LDS copy (tracked by ASYNCcnt). VDST VGPR holds the
// LDS byte address (low 32 bits of the generic pointer), VADDR the global addr.
__device__ __forceinline__ void async_ld_b128(u32 lds_addr, const void* gaddr) {
    asm volatile("global_load_async_to_lds_b128 %0, %1, off"
                 :: "v"(lds_addr), "v"(gaddr) : "memory");
}
__device__ __forceinline__ void wait_async0() {
    asm volatile("s_wait_asynccnt 0" ::: "memory");
}

// ---------------------------------------------------------------------------
// fp32 -> bf16 elementwise (plain and transposing variants)
// ---------------------------------------------------------------------------
__global__ void cvt_kernel(const float* __restrict__ src, u16* __restrict__ dst, int n) {
    int i = blockIdx.x * 256 + threadIdx.x;
    if (i < n) dst[i] = f2bf(src[i]);
}

// dst[n*K + k] = bf16(src[k*N + n])  — weights [K][N] -> [N][K]
__global__ void cvt_t_kernel(const float* __restrict__ src, u16* __restrict__ dst,
                             int K, int N) {
    int i = blockIdx.x * 256 + threadIdx.x;
    if (i >= K * N) return;
    int k = i % K, n = i / K;
    dst[i] = f2bf(src[(size_t)k * N + n]);
}

// ---------------------------------------------------------------------------
// LayerNorm over D=1024 per row; supports concat(memory, inputs) via rowsA.
// ---------------------------------------------------------------------------
__global__ __launch_bounds__(256) void ln_kernel(
    const float* __restrict__ srcA, const float* __restrict__ srcB, int rowsA,
    const float* __restrict__ g, const float* __restrict__ b,
    u16* __restrict__ out)
{
    __shared__ float rs[256], rq[256];
    int row = blockIdx.x;
    const float* src = (row < rowsA) ? (srcA + (size_t)row * DIM)
                                     : (srcB + (size_t)(row - rowsA) * DIM);
    int tid = threadIdx.x;
    float vals[4];
    float s = 0.f, sq = 0.f;
#pragma unroll
    for (int i = 0; i < 4; ++i) {
        float x = src[tid + i * 256];
        vals[i] = x; s += x; sq += x * x;
    }
    rs[tid] = s; rq[tid] = sq; __syncthreads();
    for (int off = 128; off > 0; off >>= 1) {
        if (tid < off) { rs[tid] += rs[tid + off]; rq[tid] += rq[tid + off]; }
        __syncthreads();
    }
    float mean = rs[0] * (1.0f / DIM);
    float var  = rq[0] * (1.0f / DIM) - mean * mean;
    float rstd = rsqrtf(var + 1e-5f);
#pragma unroll
    for (int i = 0; i < 4; ++i) {
        int col = tid + i * 256;
        out[(size_t)row * DIM + col] = f2bf((vals[i] - mean) * rstd * g[col] + b[col]);
    }
}

// ---------------------------------------------------------------------------
// WMMA GEMM: C[M,N] = act(A[M,K] @ B[K,N] + bias), where Bt is B pre-transposed
// to [N][K] bf16. Block = 256 threads (8 waves), tile 128x128, K-step 32.
// Both tiles staged with async global->LDS b128 copies; fragments read with
// 128-bit ds loads per the ISA VGPR layouts:
//   A (16x32): lane<16 holds K {0..7,16..23}, lane>=16 holds {8..15,24..31}
//   B (32x16): column = lane%16, K same as A: lane<16 {0..7,16..23}, lane>=16 {8..15,24..31}
// ---------------------------------------------------------------------------
__global__ __launch_bounds__(256) void gemm_bf16_kernel(
    const u16* __restrict__ A, const u16* __restrict__ Bt,
    float* __restrict__ C, const float* __restrict__ bias,
    int M, int N, int K, int act)
{
    __shared__ __align__(16) u16 As[128][32];   // [m][k]
    __shared__ __align__(16) u16 Bs[128][32];   // [n][k]
    const int tid  = threadIdx.x;
    const int wv   = tid >> 5, lane = tid & 31;
    const int hf   = lane >> 4, mr = lane & 15;
    const int m0   = blockIdx.y * 128, n0 = blockIdx.x * 128;

    const u32 asBase = (u32)(size_t)&As[0][0];
    const u32 bsBase = (u32)(size_t)&Bs[0][0];

    v8f acc[8];
#pragma unroll
    for (int i = 0; i < 8; ++i) acc[i] = vzero();

    for (int k0 = 0; k0 < K; k0 += 32) {
        // Stage A and Bt tiles: each 128 rows x 64B = 512 x 16B chunks.
#pragma unroll
        for (int i = 0; i < 2; ++i) {
            int c    = tid + i * 256;
            int row  = c >> 2, quad = c & 3;
            const char* ga = (const char*)A  + ((size_t)(m0 + row) * K + k0) * 2 + quad * 16;
            const char* gb = (const char*)Bt + ((size_t)(n0 + row) * K + k0) * 2 + quad * 16;
            async_ld_b128(asBase + row * 64 + quad * 16, ga);
            async_ld_b128(bsBase + row * 64 + quad * 16, gb);
            if (k0 + 32 < K) {                       // prefetch next K tile
                __builtin_prefetch(ga + 64, 0, 1);
                __builtin_prefetch(gb + 64, 0, 1);
            }
        }
        wait_async0();
        __syncthreads();

        Frag a;
        const u16* ap = &As[wv * 16 + mr][hf * 8];
        a.q[0] = *(const u32x4*)ap;          // K = hf*8 + 0..7
        a.q[1] = *(const u32x4*)(ap + 16);   // K = hf*8 + 16..23
#pragma unroll
        for (int n = 0; n < 8; ++n) {
            Frag bfr;
            const u16* bp = &Bs[n * 16 + mr][hf * 8];
            bfr.q[0] = *(const u32x4*)bp;        // K = hf*8 + 0..7
            bfr.q[1] = *(const u32x4*)(bp + 16); // K = hf*8 + 16..23
            acc[n] = wmma_bf16(a, bfr, acc[n]);
        }
        __syncthreads();
    }

#pragma unroll
    for (int n = 0; n < 8; ++n)
#pragma unroll
        for (int r2 = 0; r2 < 8; ++r2) {
            int m   = m0 + wv * 16 + hf * 8 + r2;  // D layout: vgpr r -> M=r(+8)
            int col = n0 + n * 16 + mr;            // N = lane%16
            float x = acc[n][r2] + (bias ? bias[col] : 0.0f);
            if (act) x = fmaxf(x, 0.0f);
            C[(size_t)m * N + col] = x;
        }
}

// ---------------------------------------------------------------------------
// Packing kernels for attention operand layouts.
// ---------------------------------------------------------------------------
__global__ void addu_pack_kernel(const float* __restrict__ qf,
                                 const float* __restrict__ u,
                                 const float* __restrict__ v,
                                 u16* __restrict__ qu, u16* __restrict__ qv, int n)
{
    int i = blockIdx.x * 256 + threadIdx.x;
    if (i >= n) return;                      // layout ((b*NH+h)*CURQ + q)*HDIM + d
    int d = i & (HDIM - 1); int t = i >> 6;
    int q = t % CURQ; t /= CURQ;
    int h = t % NH;   int b = t / NH;
    float val = qf[((size_t)q * BSZ + b) * DIM + h * HDIM + d];
    qu[i] = f2bf(val + u[h * HDIM + d]);
    qv[i] = f2bf(val + v[h * HDIM + d]);
}

// K rows kept [bh][k][d]; V stored transposed [bh][d][k] for WMMA B fragments.
__global__ void kv_pack_kernel(const float* __restrict__ kvf,
                               u16* __restrict__ kb, u16* __restrict__ vbt, int n)
{
    int i = blockIdx.x * 256 + threadIdx.x;
    if (i >= n) return;                      // i: ((b*NH+h)*FULL + k)*HDIM + d
    int d = i & (HDIM - 1); int t = i >> 6;
    int k = t % FULL; t /= FULL;
    int h = t % NH;   int b = t / NH;
    int bh = b * NH + h;
    size_t src = ((size_t)k * BSZ + b) * (2 * DIM) + h * HDIM + d;
    kb[i] = f2bf(kvf[src]);
    vbt[((size_t)bh * HDIM + d) * FULL + k] = f2bf(kvf[src + DIM]);
}

__global__ void r_pack_kernel(const float* __restrict__ rf, u16* __restrict__ rb, int n)
{
    int i = blockIdx.x * 256 + threadIdx.x;
    if (i >= n) return;                      // layout (h*FULL + j)*HDIM + d
    int d = i & (HDIM - 1); int t = i >> 6;
    int j = t % FULL; int h = t / FULL;
    rb[i] = f2bf(rf[(size_t)j * DIM + h * HDIM + d]);
}

// ---------------------------------------------------------------------------
// Fused relative attention. grid = (CURQ/16, BSZ*NH), block = 128 (4 waves).
// Dynamic LDS: Pbuf(bf16 16x1024, 32KB) + Sbuf(f32 16x1024, 64KB) = 96KB.
// rel_shift: pos[q,k] = P[q, k - q + CURQ-1]; mask: k > q + MEMR -> -inf.
// ---------------------------------------------------------------------------
__global__ __launch_bounds__(128) void attn_kernel(
    const u16* __restrict__ qu, const u16* __restrict__ qv,
    const u16* __restrict__ kb, const u16* __restrict__ vbt,
    const u16* __restrict__ rb, u16* __restrict__ outp)
{
    extern __shared__ char smem[];
    u16*   Pbuf = (u16*)smem;                       // 16*FULL bf16
    float* Sbuf = (float*)(smem + 16 * FULL * 2);   // 16*FULL f32
    u16*   Prob = Pbuf;                             // reused after combine

    const int tid = threadIdx.x;
    const int wv  = tid >> 5, lane = tid & 31;
    const int hf  = lane >> 4, mr = lane & 15;
    const int q0  = blockIdx.x * 16;
    const int bh  = blockIdx.y;
    const int b   = bh / NH, h = bh % NH;

    // ---- phase 1: P[m][j] = (query+v)[q0+m] . r[j]
    {
        Frag aq[2];
        const u16* qp = qv + ((size_t)bh * CURQ + q0 + mr) * HDIM;
#pragma unroll
        for (int s = 0; s < 2; ++s) {
            const u16* p = qp + s * 32 + hf * 8;
            aq[s].q[0] = *(const u32x4*)p;
            aq[s].q[1] = *(const u32x4*)(p + 16);
        }
        for (int jt = wv; jt < FULL / 16; jt += 4) {
            v8f acc = vzero();
#pragma unroll
            for (int s = 0; s < 2; ++s) {
                Frag bfr;
                const u16* rp = rb + ((size_t)h * FULL + jt * 16 + mr) * HDIM
                              + s * 32 + hf * 8;
                bfr.q[0] = *(const u32x4*)rp;
                bfr.q[1] = *(const u32x4*)(rp + 16);
                acc = wmma_bf16(aq[s], bfr, acc);
            }
            int col = jt * 16 + mr;
#pragma unroll
            for (int r2 = 0; r2 < 8; ++r2)
                Pbuf[(hf * 8 + r2) * FULL + col] = f2bf(acc[r2]);
        }
    }
    __syncthreads();

    // ---- phase 2: S = (Qu.K^T + shift(P)) * SCALE, causal mask
    {
        Frag aq[2];
        const u16* qp = qu + ((size_t)bh * CURQ + q0 + mr) * HDIM;
#pragma unroll
        for (int s = 0; s < 2; ++s) {
            const u16* p = qp + s * 32 + hf * 8;
            aq[s].q[0] = *(const u32x4*)p;
            aq[s].q[1] = *(const u32x4*)(p + 16);
        }
        for (int jt = wv; jt < FULL / 16; jt += 4) {
            v8f acc = vzero();
#pragma unroll
            for (int s = 0; s < 2; ++s) {
                Frag bfr;
                const u16* kp = kb + ((size_t)bh * FULL + jt * 16 + mr) * HDIM
                              + s * 32 + hf * 8;
                bfr.q[0] = *(const u32x4*)kp;
                bfr.q[1] = *(const u32x4*)(kp + 16);
                acc = wmma_bf16(aq[s], bfr, acc);
            }
            int col = jt * 16 + mr;
#pragma unroll
            for (int r2 = 0; r2 < 8; ++r2) {
                int m  = hf * 8 + r2;
                int qg = q0 + m;
                float sc;
                if (col > qg + MEMR) {
                    sc = -3.0e38f;
                } else {
                    int j = col - qg + (CURQ - 1);   // rel_shift index, in [0,FULL)
                    sc = (acc[r2] + bf2f(Pbuf[m * FULL + j])) * ATT_SCALE;
                }
                Sbuf[m * FULL + col] = sc;
            }
        }
    }
    __syncthreads();

    // ---- phase 3: softmax per row, write bf16 probs over Pbuf
    for (int row = wv; row < 16; row += 4) {
        float mx = -3.4e38f;
        for (int c = lane; c < FULL; c += 32) mx = fmaxf(mx, Sbuf[row * FULL + c]);
        for (int off = 16; off > 0; off >>= 1) mx = fmaxf(mx, __shfl_xor(mx, off, 32));
        float sum = 0.f;
        for (int c = lane; c < FULL; c += 32) {
            float e = __expf(Sbuf[row * FULL + c] - mx);
            Sbuf[row * FULL + c] = e; sum += e;
        }
        for (int off = 16; off > 0; off >>= 1) sum += __shfl_xor(sum, off, 32);
        float inv = 1.0f / sum;
        for (int c = lane; c < FULL; c += 32)
            Prob[row * FULL + c] = f2bf(Sbuf[row * FULL + c] * inv);
    }
    __syncthreads();

    // ---- phase 4: out(16 x 64) = probs(16 x 1024) @ V(1024 x 64); wave = N-tile
    {
        v8f acc = vzero();
        for (int kt = 0; kt < FULL / 32; ++kt) {
            int k0 = kt * 32;
            Frag a;
            const u16* pp = Prob + mr * FULL + k0 + hf * 8;
            a.q[0] = *(const u32x4*)pp;
            a.q[1] = *(const u32x4*)(pp + 16);
            Frag bfr;                            // V^T: [bh][d][k], K-contiguous
            const u16* vp = vbt + ((size_t)bh * HDIM + wv * 16 + mr) * FULL
                          + k0 + hf * 8;
            bfr.q[0] = *(const u32x4*)vp;
            bfr.q[1] = *(const u32x4*)(vp + 16);
            acc = wmma_bf16(a, bfr, acc);
        }
#pragma unroll
        for (int r2 = 0; r2 < 8; ++r2) {
            int q = q0 + hf * 8 + r2;
            int d = wv * 16 + mr;
            outp[((size_t)q * BSZ + b) * DIM + h * HDIM + d] = f2bf(acc[r2]);
        }
    }
}

// ---------------------------------------------------------------------------
// GRU gate elementwise pieces.
// ---------------------------------------------------------------------------
__global__ void gru_rx_kernel(const float* __restrict__ G1, const float* __restrict__ G2,
                              const float* __restrict__ x, u16* __restrict__ rx, int n)
{
    int i = blockIdx.x * 256 + threadIdx.x;
    if (i >= n) return;
    float r = 1.0f / (1.0f + __expf(-(G1[i] + G2[i])));
    rx[i] = f2bf(r * x[i]);
}

__global__ void gru_out_kernel(const float* __restrict__ G3, const float* __restrict__ G4,
                               const float* __restrict__ G5, const float* __restrict__ G6,
                               const float* __restrict__ x, const float* __restrict__ bg,
                               float* __restrict__ out, int n)
{
    int i = blockIdx.x * 256 + threadIdx.x;
    if (i >= n) return;
    float z = 1.0f / (1.0f + __expf(-(G3[i] + G4[i] - bg[i & (DIM - 1)])));
    float hc = tanhf(G5[i] + G6[i]);
    out[i] = (1.0f - z) * x[i] + z * hc;
}

// ---------------------------------------------------------------------------
// Host-side orchestration.
// ---------------------------------------------------------------------------
extern "C" void kernel_launch(void* const* d_in, const int* in_sizes, int n_in,
                              void* d_out, int out_size, void* d_ws, size_t ws_size,
                              hipStream_t stream)
{
    (void)in_sizes; (void)n_in; (void)out_size; (void)ws_size;
    const float* inputs  = (const float*)d_in[0];
    const float* memory  = (const float*)d_in[1];
    const float* pos_emb = (const float*)d_in[2];
    const float* u_in    = (const float*)d_in[3];
    const float* v_in    = (const float*)d_in[4];
    const float* W_kv    = (const float*)d_in[5];
    const float* b_kv    = (const float*)d_in[6];
    const float* W_q     = (const float*)d_in[7];
    const float* b_q     = (const float*)d_in[8];
    const float* W_proj  = (const float*)d_in[9];
    const float* b_proj  = (const float*)d_in[10];
    const float* W_pos   = (const float*)d_in[11];
    const float* b_pos   = (const float*)d_in[12];
    const float* ln1_g   = (const float*)d_in[13];
    const float* ln1_b   = (const float*)d_in[14];
    const float* ln2_g   = (const float*)d_in[15];
    const float* ln2_b   = (const float*)d_in[16];
    const float* gruW[12] = { (const float*)d_in[17], (const float*)d_in[18],
                              (const float*)d_in[19], (const float*)d_in[20],
                              (const float*)d_in[21], (const float*)d_in[22],
                              (const float*)d_in[24], (const float*)d_in[25],
                              (const float*)d_in[26], (const float*)d_in[27],
                              (const float*)d_in[28], (const float*)d_in[29] };
    const float* bg1 = (const float*)d_in[23];
    const float* bg2 = (const float*)d_in[30];
    const float* W1  = (const float*)d_in[31];
    const float* b1  = (const float*)d_in[32];
    const float* W2  = (const float*)d_in[33];
    const float* b2  = (const float*)d_in[34];
    float* out = (float*)d_out;

    char* base = (char*)d_ws;
    size_t off = 0;
    auto alloc = [&](size_t bytes) -> void* {
        void* p = base + off;
        off += (bytes + 255) & ~(size_t)255;
        return p;
    };
    auto cvt = [&](const float* s, u16* d, int n) {
        cvt_kernel<<<dim3((n + 255) / 256), 256, 0, stream>>>(s, d, n);
    };
    auto cvt_t = [&](const float* s, u16* d, int K, int N) {
        cvt_t_kernel<<<dim3((K * N + 255) / 256), 256, 0, stream>>>(s, d, K, N);
    };
    auto gemm = [&](const u16* A, const u16* Bt, float* C, const float* bias,
                    int M, int N, int K, int act) {
        gemm_bf16_kernel<<<dim3(N / 128, M / 128), 256, 0, stream>>>(A, Bt, C, bias, M, N, K, act);
    };

    // ---- bf16 weight copies (transposed to [N][K] for GEMM B operands) ------
    const float* wsrc[18] = { W_kv, W_q, W_pos, W_proj,
                              gruW[0], gruW[1], gruW[2], gruW[3], gruW[4], gruW[5],
                              gruW[6], gruW[7], gruW[8], gruW[9], gruW[10], gruW[11],
                              W1, W2 };
    const int wK[18] = { DIM, DIM, DIM, DIM, DIM, DIM, DIM, DIM, DIM, DIM,
                         DIM, DIM, DIM, DIM, DIM, DIM, DIM, HID };
    const int wN[18] = { 2 * DIM, DIM, DIM, DIM, DIM, DIM, DIM, DIM, DIM, DIM,
                         DIM, DIM, DIM, DIM, DIM, DIM, HID, DIM };
    u16* wbf[18];
    for (int i = 0; i < 18; ++i) {
        wbf[i] = (u16*)alloc((size_t)wK[i] * wN[i] * 2);
        cvt_t(wsrc[i], wbf[i], wK[i], wN[i]);
    }
    u16* pos_bf = (u16*)alloc((size_t)FULL * DIM * 2);   // A operand: plain
    cvt(pos_emb, pos_bf, FULL * DIM);

    // ---- activation workspace ----------------------------------------------
    const int TOK  = CURQ * BSZ;                 // 2048 query tokens
    const int TOKF = FULL * BSZ;                 // 4096 full tokens
    u16*   x1_bf  = (u16*)alloc((size_t)TOKF * DIM * 2);
    float* kv_f   = (float*)alloc((size_t)TOKF * 2 * DIM * 4);
    float* q_f    = (float*)alloc((size_t)TOK * DIM * 4);
    float* r_f    = (float*)alloc((size_t)FULL * DIM * 4);
    u16*   qu_bf  = (u16*)alloc((size_t)TOK * DIM * 2);
    u16*   qv_bf  = (u16*)alloc((size_t)TOK * DIM * 2);
    u16*   k_bf   = (u16*)alloc((size_t)TOKF * DIM * 2);
    u16*   v_bf   = (u16*)alloc((size_t)TOKF * DIM * 2);
    u16*   r_bf   = (u16*)alloc((size_t)NH * FULL * HDIM * 2);
    u16*   at_bf  = (u16*)alloc((size_t)TOK * DIM * 2);
    float* a1_f   = (float*)alloc((size_t)TOK * DIM * 4);
    u16*   a1_bf  = (u16*)alloc((size_t)TOK * DIM * 2);
    u16*   x_bf   = (u16*)alloc((size_t)TOK * DIM * 2);
    float* g0     = (float*)alloc((size_t)TOK * DIM * 4);
    float* g1     = (float*)alloc((size_t)TOK * DIM * 4);
    float* g2     = (float*)alloc((size_t)TOK * DIM * 4);
    float* g3     = (float*)alloc((size_t)TOK * DIM * 4);
    u16*   rx_bf  = (u16*)alloc((size_t)TOK * DIM * 2);
    float* o1_f   = (float*)alloc((size_t)TOK * DIM * 4);
    u16*   o1_bf  = (u16*)alloc((size_t)TOK * DIM * 2);
    u16*   x2_bf  = (u16*)alloc((size_t)TOK * DIM * 2);
    float* h1_f   = (float*)alloc((size_t)TOK * HID * 4);
    u16*   h1_bf  = (u16*)alloc((size_t)TOK * HID * 2);
    float* m2_f   = (float*)alloc((size_t)TOK * DIM * 4);
    u16*   m2_bf  = (u16*)alloc((size_t)TOK * DIM * 2);

    const int NTOK = TOK * DIM;                  // 2M elementwise size

    // ---- LN1 over concat(memory, inputs) -----------------------------------
    ln_kernel<<<TOKF, 256, 0, stream>>>(memory, inputs, MEMR * BSZ, ln1_g, ln1_b, x1_bf);

    // ---- projections --------------------------------------------------------
    gemm(x1_bf, wbf[0], kv_f, b_kv, TOKF, 2 * DIM, DIM, 0);                     // kv
    gemm(x1_bf + (size_t)MEMR * BSZ * DIM, wbf[1], q_f, b_q, TOK, DIM, DIM, 0); // q
    gemm(pos_bf, wbf[2], r_f, b_pos, FULL, DIM, DIM, 0);                        // r

    // ---- attention operand packing ------------------------------------------
    addu_pack_kernel<<<dim3((NTOK + 255) / 256), 256, 0, stream>>>(
        q_f, u_in, v_in, qu_bf, qv_bf, NTOK);
    kv_pack_kernel<<<dim3((TOKF * DIM + 255) / 256), 256, 0, stream>>>(
        kv_f, k_bf, v_bf, TOKF * DIM);
    r_pack_kernel<<<dim3((NH * FULL * HDIM + 255) / 256), 256, 0, stream>>>(
        r_f, r_bf, NH * FULL * HDIM);

    // ---- fused relative attention ------------------------------------------
    const int ATTN_LDS = 16 * FULL * 2 + 16 * FULL * 4;  // 96KB (of 320KB WGP LDS)
    attn_kernel<<<dim3(CURQ / 16, BSZ * NH), 128, ATTN_LDS, stream>>>(
        qu_bf, qv_bf, k_bf, v_bf, r_bf, at_bf);

    // ---- output projection + relu ------------------------------------------
    gemm(at_bf, wbf[3], a1_f, b_proj, TOK, DIM, DIM, 1);
    cvt(a1_f, a1_bf, NTOK);
    cvt(inputs, x_bf, NTOK);

    // ---- GRU gate 1: x = inputs, y = a1 ------------------------------------
    gemm(a1_bf, wbf[4], g0, nullptr, TOK, DIM, DIM, 0);    // y@Wr1
    gemm(x_bf,  wbf[5], g1, nullptr, TOK, DIM, DIM, 0);    // x@Ur1
    gru_rx_kernel<<<dim3((NTOK + 255) / 256), 256, 0, stream>>>(g0, g1, inputs, rx_bf, NTOK);
    gemm(a1_bf, wbf[6], g0, nullptr, TOK, DIM, DIM, 0);    // y@Wz1
    gemm(x_bf,  wbf[7], g1, nullptr, TOK, DIM, DIM, 0);    // x@Uz1
    gemm(a1_bf, wbf[8], g2, nullptr, TOK, DIM, DIM, 0);    // y@Wg1
    gemm(rx_bf, wbf[9], g3, nullptr, TOK, DIM, DIM, 0);    // (r*x)@Ug1
    gru_out_kernel<<<dim3((NTOK + 255) / 256), 256, 0, stream>>>(
        g0, g1, g2, g3, inputs, bg1, o1_f, NTOK);

    // ---- LN2 + MLP ----------------------------------------------------------
    ln_kernel<<<TOK, 256, 0, stream>>>(o1_f, o1_f, 0, ln2_g, ln2_b, x2_bf);
    gemm(x2_bf, wbf[16], h1_f, b1, TOK, HID, DIM, 1);      // relu(x2@W1+b1)
    cvt(h1_f, h1_bf, TOK * HID);
    gemm(h1_bf, wbf[17], m2_f, b2, TOK, DIM, HID, 1);      // relu(h@W2+b2)
    cvt(m2_f, m2_bf, NTOK);
    cvt(o1_f, o1_bf, NTOK);

    // ---- GRU gate 2: x = o1, y = m2; writes d_out ---------------------------
    gemm(m2_bf, wbf[10], g0, nullptr, TOK, DIM, DIM, 0);   // y@Wr2
    gemm(o1_bf, wbf[11], g1, nullptr, TOK, DIM, DIM, 0);   // x@Ur2
    gru_rx_kernel<<<dim3((NTOK + 255) / 256), 256, 0, stream>>>(g0, g1, o1_f, rx_bf, NTOK);
    gemm(m2_bf, wbf[12], g0, nullptr, TOK, DIM, DIM, 0);   // y@Wz2
    gemm(o1_bf, wbf[13], g1, nullptr, TOK, DIM, DIM, 0);   // x@Uz2
    gemm(m2_bf, wbf[14], g2, nullptr, TOK, DIM, DIM, 0);   // y@Wg2
    gemm(rx_bf, wbf[15], g3, nullptr, TOK, DIM, DIM, 0);   // (r*x)@Ug2
    gru_out_kernel<<<dim3((NTOK + 255) / 256), 256, 0, stream>>>(
        g0, g1, g2, g3, o1_f, bg2, out, NTOK);
}
